// MultiHeadSelfAttention_79078937854093
// MI455X (gfx1250) — matrix operand order repaired and checked by reading, not compiled, not run on hardware
//
#include <hip/hip_runtime.h>

typedef __attribute__((ext_vector_type(16))) __bf16 v16bf;
typedef __attribute__((ext_vector_type(8)))  float  v8f;

#define WMMA_BF16(A, B, C)                                                   \
  __builtin_amdgcn_wmma_f32_16x16x32_bf16(false, (A), false, (B), (short)0,  \
                                          (C), false, false)

static constexpr int kDim = 512;    // model dim
static constexpr int kHeads = 8;
static constexpr int kHd = 64;      // head dim
static constexpr int kSeq = 4096;
static constexpr int kBatch = 2;
// fold softmax scale (hd^-0.5 = 0.125) and log2(e) into Q so scores are
// directly in exp2 domain.
static constexpr float kQScale = 0.125f * 1.4426950408889634f;

// ---------------------------------------------------------------------------
// Kernel 1: qkv = x @ w_qkv^T + b_qkv, scattered into per-head bf16 buffers:
//   Q : [b,h,n,d]  (pre-scaled by kQScale)
//   K : [b,h,n,d]
//   Vt: [b,h,d,n]  (transposed so PV B-operand loads are contiguous)
// One wave = 32x64 output block (2 M-tiles x 4 N-tiles, 8 WMMAs per k-step).
// The 64-wide N block lies inside exactly one head & one q/k/v segment, so
// the scatter branch is wave-uniform.
// ---------------------------------------------------------------------------
__global__ __launch_bounds__(256) void qkv_kernel(
    const float* __restrict__ x, const float* __restrict__ w,
    const float* __restrict__ bias, __bf16* __restrict__ qb,
    __bf16* __restrict__ kb, __bf16* __restrict__ vt) {
  const int wave = (blockIdx.x * 256 + threadIdx.x) >> 5;
  const int lane = threadIdx.x & 31;
  const int tm = wave / 24;           // 256 row blocks (M = 8192, 32 each)
  const int tn = wave % 24;           // 24 col blocks (N = 1536, 64 each)
  const int m0 = tm * 32, o0 = tn * 64;
  const int seg = tn >> 3;            // 0=Q 1=K 2=V  (wave-uniform)
  const int h = tn & 7;               // head         (wave-uniform)
  const int row = lane & 15, half = lane >> 4;
  const int akb = half * 8;           // A operand K sub-offset (ISA layout)
  const int bkb = half * 8;           // B operand K sub-offset (table order, same as A)
  const float* arow0 = x + (size_t)(m0 + row) * kDim;
  const float* arow1 = arow0 + (size_t)16 * kDim;
  const float* brow = w + (size_t)(o0 + row) * kDim;  // w[o][k] row-major

  v8f acc0[4], acc1[4];
#pragma unroll
  for (int c = 0; c < 4; ++c) { v8f z = {}; acc0[c] = z; acc1[c] = z; }

  for (int k0 = 0; k0 < kDim; k0 += 32) {
    v16bf a0, a1;
#pragma unroll
    for (int i = 0; i < 8; ++i) {
      a0[i]     = (__bf16)arow0[k0 + akb + i];
      a0[i + 8] = (__bf16)arow0[k0 + 16 + akb + i];
      a1[i]     = (__bf16)arow1[k0 + akb + i];
      a1[i + 8] = (__bf16)arow1[k0 + 16 + akb + i];
    }
#pragma unroll
    for (int c = 0; c < 4; ++c) {
      v16bf b;
      const float* bp = brow + (size_t)c * 16 * kDim + k0 + bkb;
#pragma unroll
      for (int i = 0; i < 16; ++i) b[i] = (__bf16)bp[(i & 7) + ((i >> 3) << 4)];
      acc0[c] = WMMA_BF16(a0, b, acc0[c]);
      acc1[c] = WMMA_BF16(a1, b, acc1[c]);
    }
  }

#pragma unroll
  for (int c = 0; c < 4; ++c) {
    const int d = c * 16 + row;                       // col within head
    const float bv = bias[seg * 512 + h * kHd + d];
#pragma unroll
    for (int mm = 0; mm < 2; ++mm) {
      const v8f& acc = mm ? acc1[c] : acc0[c];
#pragma unroll
      for (int r = 0; r < 8; ++r) {
        const int m = m0 + mm * 16 + r + half * 8;    // C/D layout row
        const int bidx = m >> 12, n = m & 4095;
        const float val = acc[r] + bv;
        if (seg == 0) {
          qb[(size_t)((bidx * kHeads + h) * kSeq + n) * kHd + d] =
              (__bf16)(val * kQScale);
        } else if (seg == 1) {
          kb[(size_t)((bidx * kHeads + h) * kSeq + n) * kHd + d] = (__bf16)val;
        } else {
          vt[(size_t)((bidx * kHeads + h) * kHd + d) * kSeq + n] = (__bf16)val;
        }
      }
    }
  }
}

// ---------------------------------------------------------------------------
// Kernel 2: flash attention. One wave owns 16 query rows of one (b,h).
// Per 64-key block: S(16x64) = Q(16x64) Kt  -> 8 WMMAs (4 tiles x 2 k-steps)
//                   online softmax (exp2 domain; shfl reductions amortized
//                   over 4 tiles)
//                   O += P(16x64) V(64x64)  -> 8 WMMAs (2 k-steps x 4 d-tiles)
// P converted C-layout -> A-layout via per-wave LDS staging (2KB/wave).
// ---------------------------------------------------------------------------
__global__ __launch_bounds__(256) void attn_kernel(
    const __bf16* __restrict__ qb, const __bf16* __restrict__ kb,
    const __bf16* __restrict__ vt, __bf16* __restrict__ aout) {
  __shared__ __bf16 pshare[8][16 * 64];   // 2KB per wave
  const int wslot = threadIdx.x >> 5;
  const int wave = blockIdx.x * 8 + wslot;
  const int lane = threadIdx.x & 31;
  const int mt = wave & 255;              // 256 row tiles
  const int bh = wave >> 8;               // 0..15 (b*8 + h)
  const int m0 = mt * 16;
  const int row = lane & 15, half = lane >> 4;
  const int akb = half * 8, bkb = half * 8;
  const __bf16* Q = qb + (size_t)bh * kSeq * kHd;
  const __bf16* K = kb + (size_t)bh * kSeq * kHd;
  const __bf16* V = vt + (size_t)bh * kHd * kSeq;

  // Q fragments for K-dim 0..31 and 32..63 (held for whole loop)
  v16bf aq0, aq1;
  {
    const __bf16* qrow = Q + (size_t)(m0 + row) * kHd;
#pragma unroll
    for (int i = 0; i < 8; ++i) {
      aq0[i]     = qrow[akb + i];
      aq0[i + 8] = qrow[16 + akb + i];
      aq1[i]     = qrow[32 + akb + i];
      aq1[i + 8] = qrow[48 + akb + i];
    }
  }

  float mrow[8], lrow[8];
  v8f oacc[4];
#pragma unroll
  for (int r = 0; r < 8; ++r) { mrow[r] = -3.0e38f; lrow[r] = 0.f; }
#pragma unroll
  for (int t = 0; t < 4; ++t) { v8f z = {}; oacc[t] = z; }
  __bf16* pbuf = &pshare[wslot][0];

  for (int j0 = 0; j0 < kSeq; j0 += 64) {
    // --- scores for 4 key tiles ---
    v8f s[4];
#pragma unroll
    for (int kt = 0; kt < 4; ++kt) {
      v8f z = {};
      const __bf16* kp = K + (size_t)(j0 + kt * 16 + row) * kHd;
      v16bf b;
#pragma unroll
      for (int i = 0; i < 16; ++i) b[i] = kp[bkb + (i & 7) + ((i >> 3) << 4)];
      z = WMMA_BF16(aq0, b, z);
#pragma unroll
      for (int i = 0; i < 16; ++i) b[i] = kp[32 + bkb + (i & 7) + ((i >> 3) << 4)];
      s[kt] = WMMA_BF16(aq1, b, z);
    }
    // --- online softmax (base-2 domain; scale folded into Q) ---
    float alpha[8];
#pragma unroll
    for (int r = 0; r < 8; ++r) {
      float mx = fmaxf(fmaxf(s[0][r], s[1][r]), fmaxf(s[2][r], s[3][r]));
      mx = fmaxf(mx, __shfl_xor(mx, 1));
      mx = fmaxf(mx, __shfl_xor(mx, 2));
      mx = fmaxf(mx, __shfl_xor(mx, 4));
      mx = fmaxf(mx, __shfl_xor(mx, 8));   // reduce within 16-lane half
      const float mnew = fmaxf(mrow[r], mx);
      alpha[r] = exp2f(mrow[r] - mnew);
      mrow[r] = mnew;
      float ps = 0.f;
#pragma unroll
      for (int kt = 0; kt < 4; ++kt) {
        const float p = exp2f(s[kt][r] - mnew);
        s[kt][r] = p;
        ps += p;
      }
      ps += __shfl_xor(ps, 1);
      ps += __shfl_xor(ps, 2);
      ps += __shfl_xor(ps, 4);
      ps += __shfl_xor(ps, 8);
      lrow[r] = lrow[r] * alpha[r] + ps;
    }
#pragma unroll
    for (int t = 0; t < 4; ++t)
#pragma unroll
      for (int r = 0; r < 8; ++r) oacc[t][r] *= alpha[r];

    // --- P: C/D layout -> A layout through LDS ---
#pragma unroll
    for (int r = 0; r < 8; ++r) {
      const int rr = r + half * 8;
#pragma unroll
      for (int kt = 0; kt < 4; ++kt)
        pbuf[rr * 64 + kt * 16 + row] = (__bf16)s[kt][r];
    }
    __syncthreads();
    v16bf ap0, ap1;
#pragma unroll
    for (int i = 0; i < 8; ++i) {
      ap0[i]     = pbuf[row * 64 + akb + i];
      ap0[i + 8] = pbuf[row * 64 + 16 + akb + i];
      ap1[i]     = pbuf[row * 64 + 32 + akb + i];
      ap1[i + 8] = pbuf[row * 64 + 48 + akb + i];
    }
    // --- O += P * V (V stored transposed: contiguous keys per lane) ---
#pragma unroll
    for (int t = 0; t < 4; ++t) {
      const __bf16* vrow = V + (size_t)(t * 16 + row) * kSeq + j0;
      v16bf bv;
#pragma unroll
      for (int i = 0; i < 16; ++i) bv[i] = vrow[bkb + (i & 7) + ((i >> 3) << 4)];
      oacc[t] = WMMA_BF16(ap0, bv, oacc[t]);
#pragma unroll
      for (int i = 0; i < 16; ++i) bv[i] = vrow[32 + bkb + (i & 7) + ((i >> 3) << 4)];
      oacc[t] = WMMA_BF16(ap1, bv, oacc[t]);
    }
  }

  float linv[8];
#pragma unroll
  for (int r = 0; r < 8; ++r) linv[r] = 1.0f / lrow[r];
  const int b = bh >> 3, h = bh & 7;
#pragma unroll
  for (int t = 0; t < 4; ++t) {
    const int d = t * 16 + row;
#pragma unroll
    for (int r = 0; r < 8; ++r) {
      const int n = m0 + r + half * 8;
      aout[(size_t)(b * kSeq + n) * kDim + h * kHd + d] =
          (__bf16)(oacc[t][r] * linv[r]);
    }
  }
}

// ---------------------------------------------------------------------------
// Kernel 3: out = attn_out @ w_proj^T + b_proj  (fp32 output)
// One wave = 32x64 output block, 8 WMMAs per k-step.
// ---------------------------------------------------------------------------
__global__ __launch_bounds__(256) void proj_kernel(
    const __bf16* __restrict__ a, const float* __restrict__ w,
    const float* __restrict__ bias, float* __restrict__ out) {
  const int wave = (blockIdx.x * 256 + threadIdx.x) >> 5;
  const int lane = threadIdx.x & 31;
  const int tm = wave >> 3;           // 256 row blocks (M = 8192, 32 each)
  const int tn = wave & 7;            // 8 col blocks (N = 512, 64 each)
  const int m0 = tm * 32, o0 = tn * 64;
  const int row = lane & 15, half = lane >> 4;
  const int akb = half * 8, bkb = half * 8;
  const __bf16* arow0 = a + (size_t)(m0 + row) * kDim;
  const __bf16* arow1 = arow0 + (size_t)16 * kDim;
  const float* brow = w + (size_t)(o0 + row) * kDim;

  v8f acc0[4], acc1[4];
#pragma unroll
  for (int c = 0; c < 4; ++c) { v8f z = {}; acc0[c] = z; acc1[c] = z; }

  for (int k0 = 0; k0 < kDim; k0 += 32) {
    v16bf a0, a1;
#pragma unroll
    for (int i = 0; i < 8; ++i) {
      a0[i]     = arow0[k0 + akb + i];
      a0[i + 8] = arow0[k0 + 16 + akb + i];
      a1[i]     = arow1[k0 + akb + i];
      a1[i + 8] = arow1[k0 + 16 + akb + i];
    }
#pragma unroll
    for (int c = 0; c < 4; ++c) {
      v16bf b;
      const float* bp = brow + (size_t)c * 16 * kDim + k0 + bkb;
#pragma unroll
      for (int i = 0; i < 16; ++i) b[i] = (__bf16)bp[(i & 7) + ((i >> 3) << 4)];
      acc0[c] = WMMA_BF16(a0, b, acc0[c]);
      acc1[c] = WMMA_BF16(a1, b, acc1[c]);
    }
  }

#pragma unroll
  for (int c = 0; c < 4; ++c) {
    const float bb = bias[o0 + c * 16 + row];
#pragma unroll
    for (int mm = 0; mm < 2; ++mm) {
      const v8f& acc = mm ? acc1[c] : acc0[c];
#pragma unroll
      for (int r = 0; r < 8; ++r)
        out[(size_t)(m0 + mm * 16 + r + half * 8) * kDim + o0 + c * 16 + row] =
            acc[r] + bb;
    }
  }
}

// ---------------------------------------------------------------------------
extern "C" void kernel_launch(void* const* d_in, const int* in_sizes, int n_in,
                              void* d_out, int out_size, void* d_ws,
                              size_t ws_size, hipStream_t stream) {
  (void)in_sizes; (void)n_in; (void)out_size; (void)ws_size;
  const float* x      = (const float*)d_in[0];
  const float* w_qkv  = (const float*)d_in[1];
  const float* b_qkv  = (const float*)d_in[2];
  const float* w_proj = (const float*)d_in[3];
  const float* b_proj = (const float*)d_in[4];
  float* out = (float*)d_out;

  const size_t per_buf = (size_t)kBatch * kHeads * kSeq * kHd;  // 4M elems
  __bf16* qb = (__bf16*)d_ws;
  __bf16* kb = qb + per_buf;
  __bf16* vt = kb + per_buf;
  __bf16* ao = vt + per_buf;   // [B*N, C] bf16 attention output

  // 8192/32 * 1536/64 = 6144 waves -> 768 blocks of 8 waves
  qkv_kernel<<<768, 256, 0, stream>>>(x, w_qkv, b_qkv, qb, kb, vt);
  // 16 (b,h) * 256 row tiles = 4096 waves -> 512 blocks
  attn_kernel<<<512, 256, 0, stream>>>(qb, kb, vt, ao);
  // 8192/32 * 512/64 = 2048 waves -> 256 blocks
  proj_kernel<<<256, 256, 0, stream>>>(ao, w_proj, b_proj, out);
}
